// SE3Transformer_56813827391799
// MI455X (gfx1250) — matrix operand order repaired and checked by reading, not compiled, not run on hardware
//
#include <hip/hip_runtime.h>
#include <hip/hip_bf16.h>

// ---------------------------------------------------------------------------
// SE(3)-Transformer forward for MI455X (gfx1250).
// Dominant FLOPs (radial MLP GEMMs, ~14 GF) run on v_wmma_f32_16x16x32_f16
// with f32 accumulation; activations are LayerNorm'd so f16 inputs are safe.
// R tensors stored f16 in one reused 37.7MB workspace region (L2-resident).
// wave32 everywhere: shfl reductions at width 16/32, 8-wave WMMA blocks.
// Phase-3 GEMM processes two column tiles per iteration: one shared A
// fragment feeds two independent WMMA accumulator chains (ILP + A reuse).
// ---------------------------------------------------------------------------

typedef __attribute__((ext_vector_type(16))) _Float16 v16h;
typedef __attribute__((ext_vector_type(8)))  _Float16 v8h;
typedef __attribute__((ext_vector_type(8)))  float    v8f;

#define Bc   2
#define Nc   256
#define Kc   12
#define DIMc 32
#define MIDc 128
#define Ec   (Bc * Nc * Kc)     // 6144 edges
#define FBIG 3.4028234663852886e+38f

// ---------------------------------------------------------------------------
// Weight transpose + f32->f16 convert:  src[K=128][OUT] -> dst[OUT][128]
// ---------------------------------------------------------------------------
__global__ __launch_bounds__(256) void cvtT_kernel(const float* __restrict__ src,
                                                   _Float16* __restrict__ dst,
                                                   int outw) {
  int idx = blockIdx.x * 256 + threadIdx.x;
  if (idx >= outw * 128) return;
  int col = idx >> 7;          // output column
  int k   = idx & 127;         // input row
  dst[idx] = (_Float16)src[(size_t)k * outw + col];
}

// ---------------------------------------------------------------------------
// KNN: one block per (b,n); 12 rounds of argmin over 256 dists (tie: low idx)
// matches jax.lax.top_k(-masked, K) ordering.
// ---------------------------------------------------------------------------
__global__ __launch_bounds__(256) void knn_kernel(const float* __restrict__ coors,
                                                  int* __restrict__ nbrIdx,
                                                  float* __restrict__ nbrDist) {
  __shared__ float dist[256];
  __shared__ float rv[256];
  __shared__ int   ri[256];
  int node = blockIdx.x;             // b*N + n
  int b = node >> 8, n = node & 255;
  int t = threadIdx.x;
  const float* cb = coors + (size_t)b * Nc * 3;
  float x0 = cb[n * 3 + 0], y0 = cb[n * 3 + 1], z0 = cb[n * 3 + 2];
  float xt = cb[t * 3 + 0], yt = cb[t * 3 + 1], zt = cb[t * 3 + 2];
  float dx = x0 - xt, dy = y0 - yt, dz = z0 - zt;
  float d2 = dx * dx + dy * dy + dz * dz;
  float rd = (d2 > 0.f) ? sqrtf(d2) : 0.f;
  dist[t] = (t == n) ? FBIG : rd;
  __syncthreads();
  for (int kk = 0; kk < Kc; ++kk) {
    rv[t] = dist[t];
    ri[t] = t;
    __syncthreads();
    for (int s = 128; s > 0; s >>= 1) {
      if (t < s) {
        float a = rv[t], bv = rv[t + s];
        int ia = ri[t], ib = ri[t + s];
        if (bv < a || (bv == a && ib < ia)) { rv[t] = bv; ri[t] = ib; }
      }
      __syncthreads();
    }
    if (t == 0) {
      int w = ri[0];
      nbrIdx[(size_t)node * Kc + kk]  = w;
      nbrDist[(size_t)node * Kc + kk] = rv[0];
      dist[w] = FBIG;
    }
    __syncthreads();
  }
}

// ---------------------------------------------------------------------------
// A-fragment loader: row-major f16 row in LDS -> v16h in contiguous
// k order (one 16-half run at k0, k = k0 + s, matching B).
// ---------------------------------------------------------------------------
__device__ __forceinline__ v16h load_a(const _Float16* row, int k0) {
  union { v16h v; v8h h[2]; } u;
  u.h[0] = *(const v8h*)(row + k0);
  u.h[1] = *(const v8h*)(row + k0 + 8);
  return u.v;
}

// ---------------------------------------------------------------------------
// Radial MLP: 16 edges per block (8 waves).  h1 = relu(LN(rd*w1+b1)) [regs],
// h2 = relu(LN(h1@w2+b2)) via WMMA, R = h2@w3+b3 via WMMA -> f16 global.
// w2T/w3T are pre-transposed f16 [out][128].
// ---------------------------------------------------------------------------
__global__ __launch_bounds__(256) void radial_kernel(
    const float* __restrict__ nbrDist,
    const float* __restrict__ w1,  const float* __restrict__ b1,
    const float* __restrict__ g1,  const float* __restrict__ bb1,
    const _Float16* __restrict__ w2T,
    const float* __restrict__ b2,
    const float* __restrict__ g2,  const float* __restrict__ bb2,
    const _Float16* __restrict__ w3T,
    const float* __restrict__ b3,
    _Float16* __restrict__ Rout, int outw) {
  __shared__ _Float16 h1h[16][136];
  __shared__ float    h2f[16][136];
  __shared__ _Float16 h2h[16][136];

  const int tid  = threadIdx.x;
  const int wave = tid >> 5;        // 0..7
  const int lane = tid & 31;
  const int hlf  = lane >> 4;       // 0/1
  const int l16  = lane & 15;
  const size_t e0 = (size_t)blockIdx.x * 16;

  // ---- phase 1: h1 (each (wave,hlf) owns one edge; 16 lanes x 8 cols) ----
  {
    int eloc = wave * 2 + hlf;
    float rd = nbrDist[e0 + eloc];
    float v[8], s = 0.f, s2 = 0.f;
#pragma unroll
    for (int jj = 0; jj < 8; ++jj) {
      int j = l16 * 8 + jj;
      float x = rd * w1[j] + b1[j];
      v[jj] = x; s += x; s2 += x * x;
    }
#pragma unroll
    for (int m = 1; m < 16; m <<= 1) {
      s  += __shfl_xor(s,  m, 16);
      s2 += __shfl_xor(s2, m, 16);
    }
    float mu  = s * (1.f / 128.f);
    float var = s2 * (1.f / 128.f) - mu * mu;
    float rstd = rsqrtf(var + 1e-5f);
#pragma unroll
    for (int jj = 0; jj < 8; ++jj) {
      int j = l16 * 8 + jj;
      float x = (v[jj] - mu) * rstd * g1[j] + bb1[j];
      h1h[eloc][j] = (_Float16)fmaxf(x, 0.f);
    }
  }
  __syncthreads();

  // ---- phase 2: h2 = h1 @ w2 (each wave: one 16x16 col tile, K=128) ----
  {
    int n0  = wave * 16;
    int col = n0 + l16;
    v8f c = {};
#pragma unroll
    for (int kk = 0; kk < 4; ++kk) {
      v16h a = load_a(&h1h[l16][0], kk * 32 + hlf * 16);
      v16h b = *(const v16h*)(w2T + (size_t)col * 128 + kk * 32 + hlf * 16);
      c = __builtin_amdgcn_wmma_f32_16x16x32_f16(false, a, false, b,
                                                 (short)0, c, false, false);
    }
    float bias = b2[col];
    int r0 = hlf * 8;
#pragma unroll
    for (int r = 0; r < 8; ++r) h2f[r0 + r][col] = c[r] + bias;
  }
  __syncthreads();

  // ---- LN + relu over h2 (same mapping as phase 1) ----
  {
    int eloc = wave * 2 + hlf;
    float v[8], s = 0.f, s2 = 0.f;
#pragma unroll
    for (int jj = 0; jj < 8; ++jj) {
      int j = l16 * 8 + jj;
      float x = h2f[eloc][j];
      v[jj] = x; s += x; s2 += x * x;
    }
#pragma unroll
    for (int m = 1; m < 16; m <<= 1) {
      s  += __shfl_xor(s,  m, 16);
      s2 += __shfl_xor(s2, m, 16);
    }
    float mu  = s * (1.f / 128.f);
    float var = s2 * (1.f / 128.f) - mu * mu;
    float rstd = rsqrtf(var + 1e-5f);
#pragma unroll
    for (int jj = 0; jj < 8; ++jj) {
      int j = l16 * 8 + jj;
      float x = (v[jj] - mu) * rstd * g2[j] + bb2[j];
      h2h[eloc][j] = (_Float16)fmaxf(x, 0.f);
    }
  }
  __syncthreads();

  // ---- phase 3: R = h2 @ w3 + b3 ------------------------------------------
  // Two column tiles per iteration: shared A fragment, two independent WMMA
  // accumulator chains (ILP in the XDL pipe, A-fragment LDS traffic halved).
  // Wave w covers tiles {16j + 2w, 16j + 2w + 1} for all j.
  {
    int ntiles = outw >> 4;                 // 64 (OUT=1024) or 192 (OUT=3072)
    for (int tI = wave * 2; tI < ntiles; tI += 16) {
      int col0 = (tI << 4) + l16;
      int col1 = col0 + 16;
      if (tI + 16 < ntiles)                 // hint L2: this wave's next pair
        __builtin_prefetch(w3T + (size_t)(col0 + 256) * 128, 0, 1);
      v8f c0 = {}, c1 = {};
#pragma unroll
      for (int kk = 0; kk < 4; ++kk) {
        v16h a  = load_a(&h2h[l16][0], kk * 32 + hlf * 16);
        v16h bA = *(const v16h*)(w3T + (size_t)col0 * 128 + kk * 32 + hlf * 16);
        v16h bB = *(const v16h*)(w3T + (size_t)col1 * 128 + kk * 32 + hlf * 16);
        c0 = __builtin_amdgcn_wmma_f32_16x16x32_f16(false, a, false, bA,
                                                    (short)0, c0, false, false);
        c1 = __builtin_amdgcn_wmma_f32_16x16x32_f16(false, a, false, bB,
                                                    (short)0, c1, false, false);
      }
      float bias0 = b3[col0];
      float bias1 = b3[col1];
      int r0 = hlf * 8;
#pragma unroll
      for (int r = 0; r < 8; ++r) {
        Rout[(e0 + r0 + r) * (size_t)outw + col0] = (_Float16)(c0[r] + bias0);
        Rout[(e0 + r0 + r) * (size_t)outw + col1] = (_Float16)(c1[r] + bias1);
      }
    }
  }
}

// ---------------------------------------------------------------------------
// Aggregation: one wave per node; thread o handles output channel o.
// acc[o,p] += mean_k sum_{i,f} R[o,i,f] * z[i,p,f],  z[i,p,f]=sum_q ba[p,q,f]x[i,q]
// ---------------------------------------------------------------------------
template <int Di, int Do, int F>
__global__ __launch_bounds__(32) void agg_kernel(
    const int* __restrict__ nbrIdx, const _Float16* __restrict__ R,
    const float* __restrict__ basis, const float* __restrict__ xin,
    float* __restrict__ acc, int accumulate) {
  __shared__ float z[32][10];
  const int node = blockIdx.x;          // b*N + n
  const int o    = threadIdx.x;         // channel / also i for loads
  const int b    = node >> 8;
  float av[Do];
#pragma unroll
  for (int p = 0; p < Do; ++p) av[p] = 0.f;

  for (int k = 0; k < Kc; ++k) {
    size_t e = (size_t)node * Kc + k;
    int nbr = nbrIdx[e];
    size_t nb = (size_t)b * Nc + nbr;
    float x[Di];
    if (Di == 1) {
      x[0] = xin[nb * DIMc + o];
    } else {
#pragma unroll
      for (int q = 0; q < Di; ++q) x[q] = xin[(nb * DIMc + o) * Di + q];
    }
    const float* ba = basis + e * (size_t)(Do * Di * F);
#pragma unroll
    for (int p = 0; p < Do; ++p)
#pragma unroll
      for (int fi = 0; fi < F; ++fi) {
        float zz = 0.f;
#pragma unroll
        for (int q = 0; q < Di; ++q) zz += ba[(p * Di + q) * F + fi] * x[q];
        z[o][p * F + fi] = zz;
      }
    __syncthreads();
    const _Float16* Rrow = R + e * (size_t)(1024 * F) + (size_t)o * 32 * F;
#pragma unroll
    for (int p = 0; p < Do; ++p) {
      float s = 0.f;
      for (int i = 0; i < 32; ++i)
#pragma unroll
        for (int fi = 0; fi < F; ++fi)
          s += (float)Rrow[i * F + fi] * z[i][p * F + fi];
      av[p] += s;
    }
    __syncthreads();
  }
#pragma unroll
  for (int p = 0; p < Do; ++p) {
    float r = av[p] * (1.f / (float)Kc);
    size_t oi = ((size_t)node * DIMc + o) * Do + p;
    if (accumulate) acc[oi] += r; else acc[oi] = r;
  }
}

// ---------------------------------------------------------------------------
// conv_in finalize: acc0 += self0@feats, then SE3-norm both degrees -> h0,h1
// ---------------------------------------------------------------------------
__global__ __launch_bounds__(32) void norm_in_kernel(
    const float* __restrict__ acc0, const float* __restrict__ acc1,
    const float* __restrict__ feats, const float* __restrict__ self0,
    const float* __restrict__ n0g, const float* __restrict__ n0b,
    const float* __restrict__ n1g, const float* __restrict__ n1b,
    float* __restrict__ h0, float* __restrict__ h1) {
  const int node = blockIdx.x;
  const int o = threadIdx.x;
  float fo = feats[(size_t)node * DIMc + o];
  float y = 0.f;
  for (int i = 0; i < 32; ++i) y += self0[o * 32 + i] * __shfl(fo, i, 32);
  float t0 = acc0[(size_t)node * DIMc + o] + y;

  // degree 0 norm
  float nrm = fmaxf(fabsf(t0), 1e-12f);
  float phase = t0 / nrm;
  float s = nrm, s2 = nrm * nrm;
#pragma unroll
  for (int m = 1; m < 32; m <<= 1) { s += __shfl_xor(s, m, 32); s2 += __shfl_xor(s2, m, 32); }
  float mu = s * (1.f / 32.f);
  float var = s2 * (1.f / 32.f) - mu * mu;
  float ln = (nrm - mu) * rsqrtf(var + 1e-5f) * n0g[o] + n0b[o];
  h0[(size_t)node * DIMc + o] = fmaxf(ln, 0.f) * phase;

  // degree 1 norm
  float t[3];
#pragma unroll
  for (int p = 0; p < 3; ++p) t[p] = acc1[((size_t)node * DIMc + o) * 3 + p];
  float nn = sqrtf(t[0] * t[0] + t[1] * t[1] + t[2] * t[2]);
  float nrm1 = fmaxf(nn, 1e-12f);
  float s1 = nrm1, s12 = nrm1 * nrm1;
#pragma unroll
  for (int m = 1; m < 32; m <<= 1) { s1 += __shfl_xor(s1, m, 32); s12 += __shfl_xor(s12, m, 32); }
  float mu1 = s1 * (1.f / 32.f);
  float var1 = s12 * (1.f / 32.f) - mu1 * mu1;
  float ln1 = (nrm1 - mu1) * rsqrtf(var1 + 1e-5f) * n1g[o] + n1b[o];
  float tr = fmaxf(ln1, 0.f);
#pragma unroll
  for (int p = 0; p < 3; ++p)
    h1[((size_t)node * DIMc + o) * 3 + p] = tr * (t[p] / nrm1);
}

// ---------------------------------------------------------------------------
// conv_out finalize: add self terms and write outputs (deg0 then deg1).
// ---------------------------------------------------------------------------
__global__ __launch_bounds__(32) void final_kernel(
    const float* __restrict__ acc0, const float* __restrict__ acc1,
    const float* __restrict__ h0, const float* __restrict__ h1,
    const float* __restrict__ self0, const float* __restrict__ self1,
    float* __restrict__ out) {
  const int node = blockIdx.x;
  const int o = threadIdx.x;
  float h0o = h0[(size_t)node * DIMc + o];
  float y0 = 0.f;
  for (int i = 0; i < 32; ++i) y0 += self0[o * 32 + i] * __shfl(h0o, i, 32);
  out[(size_t)node * DIMc + o] = acc0[(size_t)node * DIMc + o] + y0;

  float hp[3];
#pragma unroll
  for (int p = 0; p < 3; ++p) hp[p] = h1[((size_t)node * DIMc + o) * 3 + p];
  float y[3] = {0.f, 0.f, 0.f};
  for (int i = 0; i < 32; ++i) {
    float s = self1[o * 32 + i];
#pragma unroll
    for (int p = 0; p < 3; ++p) y[p] += s * __shfl(hp[p], i, 32);
  }
#pragma unroll
  for (int p = 0; p < 3; ++p)
    out[(size_t)(Bc * Nc * DIMc) + ((size_t)node * DIMc + o) * 3 + p] =
        acc1[((size_t)node * DIMc + o) * 3 + p] + y[p];
}

// ---------------------------------------------------------------------------
// Host launcher
// ---------------------------------------------------------------------------
extern "C" void kernel_launch(void* const* d_in, const int* in_sizes, int n_in,
                              void* d_out, int out_size, void* d_ws, size_t ws_size,
                              hipStream_t stream) {
  (void)in_sizes; (void)n_in; (void)out_size; (void)ws_size;
  const float* feats = (const float*)d_in[0];
  const float* coors = (const float*)d_in[1];
  // depth-first insertion-order flattening of the params/basis dicts:
  // rad_p key order: w1,b1,ln1g,ln1b,w2,b2,ln2g,ln2b,w3,b3 (10 arrays each)
  const int pb[6] = {2, 12, 27, 37, 47, 57};   // in00,in01,out00,out01,out10,out11
  const int Fp[6] = {1, 1, 1, 1, 1, 3};
  const int SELF_IN0 = 22, N0G = 23, N0B = 24, N1G = 25, N1B = 26;
  const int SELF_OUT0 = 67, SELF_OUT1 = 68;
  const int BAS[4] = {69, 70, 71, 72};         // '0,0','0,1','1,0','1,1'

  // workspace carve-out (~40.6 MB total)
  size_t cur = 0;
  char* wsb = (char*)d_ws;
  auto take = [&](size_t bytes) -> void* {
    void* p = wsb + cur;
    cur = (cur + bytes + 255) & ~(size_t)255;
    return p;
  };
  int*      nbrIdx  = (int*)take((size_t)Ec * sizeof(int));
  float*    nbrDist = (float*)take((size_t)Ec * sizeof(float));
  _Float16* w2T[6];
  _Float16* w3T[6];
  for (int i = 0; i < 6; ++i) w2T[i] = (_Float16*)take((size_t)128 * 128 * sizeof(_Float16));
  for (int i = 0; i < 6; ++i) w3T[i] = (_Float16*)take((size_t)1024 * Fp[i] * 128 * sizeof(_Float16));
  float* acc0 = (float*)take((size_t)Bc * Nc * DIMc * sizeof(float));
  float* acc1 = (float*)take((size_t)Bc * Nc * DIMc * 3 * sizeof(float));
  float* h0   = (float*)take((size_t)Bc * Nc * DIMc * sizeof(float));
  float* h1   = (float*)take((size_t)Bc * Nc * DIMc * 3 * sizeof(float));
  _Float16* Rbuf = (_Float16*)take((size_t)Ec * 3072 * sizeof(_Float16));  // reused per pair

  // 1) weight transpose + f16 convert (all pairs up front; L2-resident)
  for (int i = 0; i < 6; ++i) {
    const float* w2 = (const float*)d_in[pb[i] + 4];
    const float* w3 = (const float*)d_in[pb[i] + 8];
    int outw = 1024 * Fp[i];
    cvtT_kernel<<<dim3((128 * 128 + 255) / 256), dim3(256), 0, stream>>>(w2, w2T[i], 128);
    cvtT_kernel<<<dim3((outw * 128 + 255) / 256), dim3(256), 0, stream>>>(w3, w3T[i], outw);
  }

  // 2) KNN
  knn_kernel<<<dim3(Bc * Nc), dim3(256), 0, stream>>>(coors, nbrIdx, nbrDist);

  auto radial = [&](int i) {
    radial_kernel<<<dim3(Ec / 16), dim3(256), 0, stream>>>(
        nbrDist,
        (const float*)d_in[pb[i] + 0], (const float*)d_in[pb[i] + 1],
        (const float*)d_in[pb[i] + 2], (const float*)d_in[pb[i] + 3],
        w2T[i],
        (const float*)d_in[pb[i] + 5], (const float*)d_in[pb[i] + 6],
        (const float*)d_in[pb[i] + 7],
        w3T[i],
        (const float*)d_in[pb[i] + 9],
        Rbuf, 1024 * Fp[i]);
  };

  // 3) conv_in: pairs (0,0) -> acc0, (0,1) -> acc1
  radial(0);
  agg_kernel<1, 1, 1><<<dim3(Bc * Nc), dim3(32), 0, stream>>>(
      nbrIdx, Rbuf, (const float*)d_in[BAS[0]], feats, acc0, 0);
  radial(1);
  agg_kernel<1, 3, 1><<<dim3(Bc * Nc), dim3(32), 0, stream>>>(
      nbrIdx, Rbuf, (const float*)d_in[BAS[1]], feats, acc1, 0);

  // 4) self + SE3 norm -> h0, h1
  norm_in_kernel<<<dim3(Bc * Nc), dim3(32), 0, stream>>>(
      acc0, acc1, feats, (const float*)d_in[SELF_IN0],
      (const float*)d_in[N0G], (const float*)d_in[N0B],
      (const float*)d_in[N1G], (const float*)d_in[N1B], h0, h1);

  // 5) conv_out: (0,0)->acc0=, (0,1)->acc1=, (1,0)->acc0+=, (1,1)->acc1+=
  radial(2);
  agg_kernel<1, 1, 1><<<dim3(Bc * Nc), dim3(32), 0, stream>>>(
      nbrIdx, Rbuf, (const float*)d_in[BAS[0]], h0, acc0, 0);
  radial(3);
  agg_kernel<1, 3, 1><<<dim3(Bc * Nc), dim3(32), 0, stream>>>(
      nbrIdx, Rbuf, (const float*)d_in[BAS[1]], h0, acc1, 0);
  radial(4);
  agg_kernel<3, 1, 1><<<dim3(Bc * Nc), dim3(32), 0, stream>>>(
      nbrIdx, Rbuf, (const float*)d_in[BAS[2]], h1, acc0, 1);
  radial(5);
  agg_kernel<3, 3, 3><<<dim3(Bc * Nc), dim3(32), 0, stream>>>(
      nbrIdx, Rbuf, (const float*)d_in[BAS[3]], h1, acc1, 1);

  // 6) final: add self terms, write outputs
  final_kernel<<<dim3(Bc * Nc), dim3(32), 0, stream>>>(
      acc0, acc1, h0, h1,
      (const float*)d_in[SELF_OUT0], (const float*)d_in[SELF_OUT1],
      (float*)d_out);
}
